// LSTM_26826365730936
// MI455X (gfx1250) — matrix operand order repaired and checked by reading, not compiled, not run on hardware
//
#include <hip/hip_runtime.h>
#include <hip/hip_bf16.h>
#include <math.h>

typedef __attribute__((ext_vector_type(16))) __bf16 v16bf;
typedef __attribute__((ext_vector_type(8)))  float  v8f;

#define BB   128
#define TT   256
#define SS   64
#define INW  3
#define HH   512
#define KK   10
#define CSLW 60
#define OUTW 121
#define G4H  2048      // 4*H
#define KIN1 64        // padded IN+CSL (63 -> 64)
#define KIN2 576       // padded IN+CSL+H (575 -> 576, zero col at 63)

__device__ __forceinline__ unsigned short f2bf(float f) {
  unsigned u = __float_as_uint(f);
  return (unsigned short)((u + 0x7FFFu + ((u >> 16) & 1u)) >> 16);
}
__device__ __forceinline__ float sigf(float x) { return 1.0f / (1.0f + __expf(-x)); }

// ---------------------------------------------------------------- init state
__global__ void init_state(const float* h1, const float* c1, const float* h2,
                           const float* c2, float* h1f, float* c1f, float* h2f,
                           float* c2f, unsigned short* h1b, unsigned short* h2b,
                           float* p) {
  int i = blockIdx.x * blockDim.x + threadIdx.x;
  if (i < BB * HH) {
    float a = h1[i], b = c1[i], c = h2[i], d = c2[i];
    h1f[i] = a; c1f[i] = b; h2f[i] = c; c2f[i] = d;
    h1b[i] = f2bf(a); h2b[i] = f2bf(c);
  }
  if (i < BB * 32) p[i] = 0.0f;
}

// ------------------------------------------------------- pack x into feat bf16
__global__ void conv_x(const float* x, unsigned short* feat) {
  int i = blockIdx.x * blockDim.x + threadIdx.x;   // T*B
  if (i >= TT * BB) return;
  int t = i / BB, b = i % BB;
  unsigned short* f = feat + (size_t)(t * BB + b) * KIN2;
  const float* xp = x + (size_t)b * TT * INW + (size_t)t * INW;
  f[0] = f2bf(xp[0]); f[1] = f2bf(xp[1]); f[2] = f2bf(xp[2]);
  f[63] = 0;
}

// ----------------------------------------------- weight convert with padding
// pad63: insert a zero column at k==63 (cols >63 shift down by one in src)
__global__ void conv_w(const float* src, unsigned short* dst, int Nsrc, int Ksrc,
                       int Kdst, int pad63, int total) {
  int i = blockIdx.x * blockDim.x + threadIdx.x;
  if (i >= total) return;
  int n = i / Kdst, k = i % Kdst;
  float v = 0.0f;
  if (pad63) {
    if (k != 63) {
      int ks = (k < 63) ? k : k - 1;
      if (ks < Ksrc && n < Nsrc) v = src[(size_t)n * Ksrc + ks];
    }
  } else {
    if (k < Ksrc && n < Nsrc) v = src[(size_t)n * Ksrc + k];
  }
  dst[i] = f2bf(v);
}

// --------------------------------------------------- attention window (f32)
__global__ void attn_kernel(int t, const float* h1f, const float* W1,
                            const float* b1, const float* sent, float* p,
                            unsigned short* feat) {
  int b = blockIdx.x;
  int tid = threadIdx.x;            // 256 threads
  __shared__ float part[30][8];
  __shared__ float pv[30];
  __shared__ float phi[64];
  const float* hrow = h1f + (size_t)b * HH;
  if (tid < 240) {
    int j = tid >> 3, q = tid & 7;
    const float* w = W1 + (size_t)j * HH + q * 64;
    const float* hh = hrow + q * 64;
    float s = 0.f;
    #pragma unroll 8
    for (int k = 0; k < 64; ++k) s += hh[k] * w[k];
    part[j][q] = s;
  }
  __syncthreads();
  if (tid < 30) {
    float s = b1[tid];
    #pragma unroll
    for (int q = 0; q < 8; ++q) s += part[tid][q];
    float ip = __expf(s);
    float pn = (tid < 20) ? ip : (ip - p[b * 32 + tid]);  // kappa quirk
    p[b * 32 + tid] = pn;
    pv[tid] = pn;
  }
  __syncthreads();
  if (tid < SS) {
    float u = (float)(tid + 1);
    float s = 0.f;
    #pragma unroll
    for (int k = 0; k < KK; ++k) {
      float d = pv[20 + k] - u;
      s += pv[k] * __expf(-pv[10 + k] * d * d);
    }
    phi[tid] = s;
  }
  __syncthreads();
  if (tid < CSLW) {
    const float* sb = sent + (size_t)b * SS * CSLW + tid;
    float s = 0.f;
    for (int si = 0; si < SS; ++si) s += phi[si] * sb[si * CSLW];
    feat[(size_t)(t * BB + b) * KIN2 + 3 + tid] = f2bf(s);
  }
}

// ------------------------------- fused LSTM step: WMMA gates + pointwise h/c
// Each wave computes TWO 16-row M tiles sharing one set of weight fragments:
// D[m, g*H+n] = bias + Ain[m,:kin] @ Wi[g*H+n,:]^T + Hb[m,:512] @ Wh[g*H+n,:]^T
__global__ void __launch_bounds__(256) lstm_step(
    const unsigned short* Ain, int astride, int kin,   // [B, astride] use kin cols
    const unsigned short* Wi,                          // [2048, kin] bf16
    const unsigned short* Hb,                          // [B, 512] prev h bf16
    const unsigned short* Wh,                          // [2048, 512] bf16
    const float* bias, float* cst, float* hf32,
    unsigned short* hb_out, unsigned short* hextra, int estride) {
  const int lane = threadIdx.x & 31;
  const int wave = threadIdx.x >> 5;
  const int tile = blockIdx.x * 8 + wave;   // 128 waves: 4 (M-pair) x 32 (N)
  const int ni = tile & 31;
  const int mp = tile >> 5;                 // 0..3, each covers 32 batch rows
  const int l15 = lane & 15;
  const int hi = lane >> 4;

  const int m0 = mp * 32 + l15;  // A row, first M tile
  const int m1 = m0 + 16;        // A row, second M tile
  const int n = ni * 16 + l15;   // B-fragment / output column (within H)

  v8f a00, a01, a02, a03;   // gates i,f,g,o for M tile 0
  v8f a10, a11, a12, a13;   // gates i,f,g,o for M tile 1
  {
    float b0 = bias[0 * HH + n], b1v = bias[1 * HH + n];
    float b2v = bias[2 * HH + n], b3 = bias[3 * HH + n];
    a00 = (v8f){b0, b0, b0, b0, b0, b0, b0, b0};
    a01 = (v8f){b1v, b1v, b1v, b1v, b1v, b1v, b1v, b1v};
    a02 = (v8f){b2v, b2v, b2v, b2v, b2v, b2v, b2v, b2v};
    a03 = (v8f){b3, b3, b3, b3, b3, b3, b3, b3};
    a10 = a00; a11 = a01; a12 = a02; a13 = a03;
  }

  union Frag { v16bf v; uint4 q[2]; };
  Frag af0, af1, bf0, bf1, bf2, bf3;

  // phase 1: time-step input part (kin cols of Ain) x Wi
  const unsigned short* arow0 = Ain + (size_t)m0 * astride;
  const unsigned short* arow1 = Ain + (size_t)m1 * astride;
  for (int k0 = 0; k0 < kin; k0 += 32) {
    int ka = k0 + hi * 16;
    af0.q[0] = *(const uint4*)(arow0 + ka);
    af0.q[1] = *(const uint4*)(arow0 + ka + 8);
    af1.q[0] = *(const uint4*)(arow1 + ka);
    af1.q[1] = *(const uint4*)(arow1 + ka + 8);
    const unsigned short* w0 = Wi + (size_t)(0 * HH + n) * kin + k0 + hi * 16;
    const unsigned short* w1 = Wi + (size_t)(1 * HH + n) * kin + k0 + hi * 16;
    const unsigned short* w2 = Wi + (size_t)(2 * HH + n) * kin + k0 + hi * 16;
    const unsigned short* w3 = Wi + (size_t)(3 * HH + n) * kin + k0 + hi * 16;
    bf0.q[0] = *(const uint4*)(w0); bf0.q[1] = *(const uint4*)(w0 + 8);
    bf1.q[0] = *(const uint4*)(w1); bf1.q[1] = *(const uint4*)(w1 + 8);
    bf2.q[0] = *(const uint4*)(w2); bf2.q[1] = *(const uint4*)(w2 + 8);
    bf3.q[0] = *(const uint4*)(w3); bf3.q[1] = *(const uint4*)(w3 + 8);
    a00 = __builtin_amdgcn_wmma_f32_16x16x32_bf16(false, af0.v, false, bf0.v, (short)0, a00, false, false);
    a10 = __builtin_amdgcn_wmma_f32_16x16x32_bf16(false, af1.v, false, bf0.v, (short)0, a10, false, false);
    a01 = __builtin_amdgcn_wmma_f32_16x16x32_bf16(false, af0.v, false, bf1.v, (short)0, a01, false, false);
    a11 = __builtin_amdgcn_wmma_f32_16x16x32_bf16(false, af1.v, false, bf1.v, (short)0, a11, false, false);
    a02 = __builtin_amdgcn_wmma_f32_16x16x32_bf16(false, af0.v, false, bf2.v, (short)0, a02, false, false);
    a12 = __builtin_amdgcn_wmma_f32_16x16x32_bf16(false, af1.v, false, bf2.v, (short)0, a12, false, false);
    a03 = __builtin_amdgcn_wmma_f32_16x16x32_bf16(false, af0.v, false, bf3.v, (short)0, a03, false, false);
    a13 = __builtin_amdgcn_wmma_f32_16x16x32_bf16(false, af1.v, false, bf3.v, (short)0, a13, false, false);
  }
  // phase 2: recurrent part h_{t-1} (512) x Wh
  const unsigned short* hrow0 = Hb + (size_t)m0 * HH;
  const unsigned short* hrow1 = Hb + (size_t)m1 * HH;
  for (int k0 = 0; k0 < HH; k0 += 32) {
    int ka = k0 + hi * 16;
    af0.q[0] = *(const uint4*)(hrow0 + ka);
    af0.q[1] = *(const uint4*)(hrow0 + ka + 8);
    af1.q[0] = *(const uint4*)(hrow1 + ka);
    af1.q[1] = *(const uint4*)(hrow1 + ka + 8);
    const unsigned short* w0 = Wh + (size_t)(0 * HH + n) * HH + k0 + hi * 16;
    const unsigned short* w1 = Wh + (size_t)(1 * HH + n) * HH + k0 + hi * 16;
    const unsigned short* w2 = Wh + (size_t)(2 * HH + n) * HH + k0 + hi * 16;
    const unsigned short* w3 = Wh + (size_t)(3 * HH + n) * HH + k0 + hi * 16;
    bf0.q[0] = *(const uint4*)(w0); bf0.q[1] = *(const uint4*)(w0 + 8);
    bf1.q[0] = *(const uint4*)(w1); bf1.q[1] = *(const uint4*)(w1 + 8);
    bf2.q[0] = *(const uint4*)(w2); bf2.q[1] = *(const uint4*)(w2 + 8);
    bf3.q[0] = *(const uint4*)(w3); bf3.q[1] = *(const uint4*)(w3 + 8);
    a00 = __builtin_amdgcn_wmma_f32_16x16x32_bf16(false, af0.v, false, bf0.v, (short)0, a00, false, false);
    a10 = __builtin_amdgcn_wmma_f32_16x16x32_bf16(false, af1.v, false, bf0.v, (short)0, a10, false, false);
    a01 = __builtin_amdgcn_wmma_f32_16x16x32_bf16(false, af0.v, false, bf1.v, (short)0, a01, false, false);
    a11 = __builtin_amdgcn_wmma_f32_16x16x32_bf16(false, af1.v, false, bf1.v, (short)0, a11, false, false);
    a02 = __builtin_amdgcn_wmma_f32_16x16x32_bf16(false, af0.v, false, bf2.v, (short)0, a02, false, false);
    a12 = __builtin_amdgcn_wmma_f32_16x16x32_bf16(false, af1.v, false, bf2.v, (short)0, a12, false, false);
    a03 = __builtin_amdgcn_wmma_f32_16x16x32_bf16(false, af0.v, false, bf3.v, (short)0, a03, false, false);
    a13 = __builtin_amdgcn_wmma_f32_16x16x32_bf16(false, af1.v, false, bf3.v, (short)0, a13, false, false);
  }

  union AccU { v8f v; float f[8]; };
  AccU gi, gf, gg, go;
  #pragma unroll
  for (int half = 0; half < 2; ++half) {
    if (half == 0) { gi.v = a00; gf.v = a01; gg.v = a02; go.v = a03; }
    else           { gi.v = a10; gf.v = a11; gg.v = a12; go.v = a13; }
    const int mbase = mp * 32 + half * 16 + hi * 8;  // C/D: M = j + 8*(lane>=16)
    #pragma unroll
    for (int j = 0; j < 8; ++j) {
      int mo = mbase + j;
      size_t idx = (size_t)mo * HH + n;
      float cold = cst[idx];
      float cn = sigf(gf.f[j]) * cold + sigf(gi.f[j]) * tanhf(gg.f[j]);
      float hn = sigf(go.f[j]) * tanhf(cn);
      cst[idx] = cn;
      hf32[idx] = hn;
      unsigned short hb = f2bf(hn);
      hb_out[idx] = hb;
      hextra[(size_t)mo * estride + n] = hb;
    }
  }
}

// ------------------------------------------------ final projection y @ W2^T
__global__ void __launch_bounds__(256) out_gemm(const unsigned short* Y,
                                                const unsigned short* W2p,
                                                const float* b2, float* out) {
  const int lane = threadIdx.x & 31;
  const int wave = threadIdx.x >> 5;
  const int tile = blockIdx.x * 8 + wave;   // 16384 tiles: 2048 (M) x 8 (N)
  const int ni = tile & 7;
  const int mi = tile >> 3;
  const int l15 = lane & 15;
  const int hi = lane >> 4;
  const int n = ni * 16 + l15;
  const int m = mi * 16 + l15;
  float bv = (n < OUTW) ? b2[n] : 0.0f;
  v8f acc = (v8f){bv, bv, bv, bv, bv, bv, bv, bv};
  union Frag { v16bf v; uint4 q[2]; };
  Frag af, bf;
  const unsigned short* yrow = Y + (size_t)m * HH;
  const unsigned short* wrow = W2p + (size_t)n * HH;
  for (int k0 = 0; k0 < HH; k0 += 32) {
    af.q[0] = *(const uint4*)(yrow + k0 + hi * 16);
    af.q[1] = *(const uint4*)(yrow + k0 + hi * 16 + 8);
    bf.q[0] = *(const uint4*)(wrow + k0 + hi * 16);
    bf.q[1] = *(const uint4*)(wrow + k0 + hi * 16 + 8);
    acc = __builtin_amdgcn_wmma_f32_16x16x32_bf16(false, af.v, false, bf.v, (short)0, acc, false, false);
  }
  union AccU { v8f v; float f[8]; };
  AccU r; r.v = acc;
  if (n < OUTW) {
    #pragma unroll
    for (int j = 0; j < 8; ++j) {
      int mo = mi * 16 + hi * 8 + j;       // Y-row = t*B + b
      int orow = (mo & (BB - 1)) * TT + (mo >> 7);  // reference order b*T + t
      out[(size_t)orow * OUTW + n] = r.f[j];
    }
  }
}

// ------------------------------------------------------- final state copies
__global__ void copy_states(const float* h1f, const float* c1f, const float* h2f,
                            const float* c2f, float* dst) {
  int i = blockIdx.x * blockDim.x + threadIdx.x;
  if (i >= BB * HH) return;
  dst[i] = h1f[i];
  dst[BB * HH + i] = c1f[i];
  dst[2 * BB * HH + i] = h2f[i];
  dst[3 * BB * HH + i] = c2f[i];
}

extern "C" void kernel_launch(void* const* d_in, const int* in_sizes, int n_in,
                              void* d_out, int out_size, void* d_ws, size_t ws_size,
                              hipStream_t stream) {
  (void)in_sizes; (void)n_in; (void)out_size; (void)ws_size;
  const float* x    = (const float*)d_in[0];
  const float* sent = (const float*)d_in[1];
  const float* h1_0 = (const float*)d_in[2];
  const float* c1_0 = (const float*)d_in[3];
  const float* h2_0 = (const float*)d_in[4];
  const float* c2_0 = (const float*)d_in[5];
  const float* Wih1 = (const float*)d_in[6];
  const float* Whh1 = (const float*)d_in[7];
  const float* bc   = (const float*)d_in[8];
  const float* Wih2 = (const float*)d_in[9];
  const float* Whh2 = (const float*)d_in[10];
  const float* bl   = (const float*)d_in[11];
  const float* W1   = (const float*)d_in[12];
  const float* b1   = (const float*)d_in[13];
  const float* W2   = (const float*)d_in[14];
  const float* b2   = (const float*)d_in[15];
  float* out = (float*)d_out;

  char* ws = (char*)d_ws;
  size_t off = 0;
  auto alloc = [&](size_t bytes) -> char* {
    char* pp = ws + off;
    off = (off + bytes + 255) & ~(size_t)255;
    return pp;
  };
  unsigned short* feat = (unsigned short*)alloc((size_t)TT * BB * KIN2 * 2);
  unsigned short* Y    = (unsigned short*)alloc((size_t)TT * BB * HH * 2);
  unsigned short* Wi1b = (unsigned short*)alloc((size_t)G4H * KIN1 * 2);
  unsigned short* Wh1b = (unsigned short*)alloc((size_t)G4H * HH * 2);
  unsigned short* Wi2b = (unsigned short*)alloc((size_t)G4H * KIN2 * 2);
  unsigned short* Wh2b = (unsigned short*)alloc((size_t)G4H * HH * 2);
  unsigned short* W2pb = (unsigned short*)alloc((size_t)128 * HH * 2);
  float* h1f = (float*)alloc((size_t)BB * HH * 4);
  float* c1f = (float*)alloc((size_t)BB * HH * 4);
  float* h2f = (float*)alloc((size_t)BB * HH * 4);
  float* c2f = (float*)alloc((size_t)BB * HH * 4);
  unsigned short* h1b0 = (unsigned short*)alloc((size_t)BB * HH * 2);
  unsigned short* h1b1 = (unsigned short*)alloc((size_t)BB * HH * 2);
  unsigned short* h2b0 = (unsigned short*)alloc((size_t)BB * HH * 2);
  unsigned short* h2b1 = (unsigned short*)alloc((size_t)BB * HH * 2);
  float* pbuf = (float*)alloc((size_t)BB * 32 * 4);

  init_state<<<(BB * HH + 255) / 256, 256, 0, stream>>>(h1_0, c1_0, h2_0, c2_0,
                                                        h1f, c1f, h2f, c2f,
                                                        h1b0, h2b0, pbuf);
  conv_x<<<(TT * BB + 255) / 256, 256, 0, stream>>>(x, feat);
  conv_w<<<(G4H * KIN1 + 255) / 256, 256, 0, stream>>>(Wih1, Wi1b, G4H, 63, KIN1, 1, G4H * KIN1);
  conv_w<<<(G4H * HH + 255) / 256, 256, 0, stream>>>(Whh1, Wh1b, G4H, HH, HH, 0, G4H * HH);
  conv_w<<<(G4H * KIN2 + 255) / 256, 256, 0, stream>>>(Wih2, Wi2b, G4H, 575, KIN2, 1, G4H * KIN2);
  conv_w<<<(G4H * HH + 255) / 256, 256, 0, stream>>>(Whh2, Wh2b, G4H, HH, HH, 0, G4H * HH);
  conv_w<<<(128 * HH + 255) / 256, 256, 0, stream>>>(W2, W2pb, OUTW, HH, HH, 0, 128 * HH);

  unsigned short* h1pp[2] = {h1b0, h1b1};
  unsigned short* h2pp[2] = {h2b0, h2b1};

  // LSTM1 with attention window, sequential over T
  for (int t = 0; t < TT; ++t) {
    attn_kernel<<<BB, 256, 0, stream>>>(t, h1f, W1, b1, sent, pbuf, feat);
    lstm_step<<<16, 256, 0, stream>>>(feat + (size_t)t * BB * KIN2, KIN2, KIN1,
                                      Wi1b, h1pp[t & 1], Wh1b, bc, c1f, h1f,
                                      h1pp[(t + 1) & 1],
                                      feat + (size_t)t * BB * KIN2 + KIN1, KIN2);
  }
  // LSTM2, sequential over T
  for (int t = 0; t < TT; ++t) {
    lstm_step<<<16, 256, 0, stream>>>(feat + (size_t)t * BB * KIN2, KIN2, KIN2,
                                      Wi2b, h2pp[t & 1], Wh2b, bl, c2f, h2f,
                                      h2pp[(t + 1) & 1],
                                      Y + (size_t)t * BB * HH, HH);
  }
  out_gemm<<<2048, 256, 0, stream>>>(Y, W2pb, b2, out);
  copy_states<<<(BB * HH + 255) / 256, 256, 0, stream>>>(h1f, c1f, h2f, c2f,
                                                         out + (size_t)BB * TT * OUTW);
}
